// CRF_32856499815008
// MI455X (gfx1250) — matrix operand order repaired and checked by reading, not compiled, not run on hardware
//
#include <hip/hip_runtime.h>

#define B_ 32
#define T_ 128
#define L_ 48
#define V_ 10000

typedef __attribute__((ext_vector_type(16))) _Float16 v16h;
typedef __attribute__((ext_vector_type(8)))  float    v8f;

// Pack 16 f32 (4x float4) into a v16h fragment (f32 -> f16 cvt, lowers to v_cvt_pk_f16_f32).
__device__ __forceinline__ v16h pack16(float4 g0, float4 g1, float4 g2, float4 g3) {
  v16h r;
  r[0]  = (_Float16)g0.x; r[1]  = (_Float16)g0.y; r[2]  = (_Float16)g0.z; r[3]  = (_Float16)g0.w;
  r[4]  = (_Float16)g1.x; r[5]  = (_Float16)g1.y; r[6]  = (_Float16)g1.z; r[7]  = (_Float16)g1.w;
  r[8]  = (_Float16)g2.x; r[9]  = (_Float16)g2.y; r[10] = (_Float16)g2.z; r[11] = (_Float16)g2.w;
  r[12] = (_Float16)g3.x; r[13] = (_Float16)g3.y; r[14] = (_Float16)g3.z; r[15] = (_Float16)g3.w;
  return r;
}

// -----------------------------------------------------------------------------
// Phase 0: one-time f32 -> f16 conversion of e_feats (48 x 10000 = 480K elems).
// Removes 12 b128 loads + 24 packed cvts per chunk from the hot GEMM loop.
// -----------------------------------------------------------------------------
__global__ void cvt_e_f16(const float* __restrict__ ef, _Float16* __restrict__ ef16, int n) {
  const int i = blockIdx.x * blockDim.x + threadIdx.x;
  if (i < n) ef16[i] = (_Float16)ef[i];
}

// -----------------------------------------------------------------------------
// Phase 1: em[b*T+t][l] = sum_v X[b,t,v] * E[l,v]   (M=4096, N=48, K=10000)
// One wave per 16-row M-tile; 3 N-tiles of 16 accumulated in v8f each.
// K loop: 312 full chunks of 32 + one 16-wide tail that lands exactly on the
// lane-half boundary (hi-half K groups fully OOB -> zero fragments).
// -----------------------------------------------------------------------------
__global__ __launch_bounds__(256) void emission_gemm(
    const float* __restrict__ X, const _Float16* __restrict__ E16,
    float* __restrict__ em) {
  const int wave = threadIdx.x >> 5;
  const int lane = threadIdx.x & 31;
  const int mt   = blockIdx.x * 8 + wave;   // 0..255 (M tile)
  const int m    = lane & 15;               // A row within tile / B col within tile
  const int hi   = lane >> 4;               // lane half
  const int ka   = hi * 16;                 // A K sub-base within chunk
  const int kbb  = hi * 16;                 // B K sub-base within chunk

  const float*    Xrow = X   + (size_t)(mt * 16 + m) * V_;
  const _Float16* E0   = E16 + (size_t)(0 * 16 + m) * V_ + kbb;
  const _Float16* E1   = E16 + (size_t)(1 * 16 + m) * V_ + kbb;
  const _Float16* E2   = E16 + (size_t)(2 * 16 + m) * V_ + kbb;

  v8f c0 = {}; v8f c1 = {}; v8f c2 = {};

  for (int kc = 0; kc < 312; ++kc) {
    const int k0 = kc * 32;
    // Stream-prefetch X one chunk group ahead (global_prefetch_b8, speculative).
    __builtin_prefetch(Xrow + k0 + 1024, 0, 1);

    const float4* pa0 = (const float4*)(Xrow + k0 + ka);
    const float4* pa1 = (const float4*)(Xrow + k0 + ka + 8);
    v16h a  = pack16(pa0[0], pa0[1], pa1[0], pa1[1]);
    v16h b0 = *(const v16h*)(E0 + k0);   // 32B of f16, 32B-aligned (2x b128)
    v16h b1 = *(const v16h*)(E1 + k0);
    v16h b2 = *(const v16h*)(E2 + k0);

    c0 = __builtin_amdgcn_wmma_f32_16x16x32_f16(false, a, false, b0, (short)0, c0, false, false);
    c1 = __builtin_amdgcn_wmma_f32_16x16x32_f16(false, a, false, b1, (short)0, c1, false, false);
    c2 = __builtin_amdgcn_wmma_f32_16x16x32_f16(false, a, false, b2, (short)0, c2, false, false);
  }

  { // Tail: k0 = 9984, only K = 9984..9999 valid.
    const int k0 = 9984;
    const float4 z4 = make_float4(0.f, 0.f, 0.f, 0.f);
    const float4* pa0 = (const float4*)(Xrow + k0);        // K=9984..9999 : valid (lo half)
    v16h a = (hi == 0) ? pack16(pa0[0], pa0[1], pa0[2], pa0[3]) : pack16(z4, z4, z4, z4); // hi half OOB -> 0
    v16h b0 = {}, b1 = {}, b2 = {};
    if (hi == 0) {                                         // lo half K valid, hi half OOB
      b0 = *(const v16h*)(E0 + k0);
      b1 = *(const v16h*)(E1 + k0);
      b2 = *(const v16h*)(E2 + k0);
    }
    c0 = __builtin_amdgcn_wmma_f32_16x16x32_f16(false, a, false, b0, (short)0, c0, false, false);
    c1 = __builtin_amdgcn_wmma_f32_16x16x32_f16(false, a, false, b1, (short)0, c1, false, false);
    c2 = __builtin_amdgcn_wmma_f32_16x16x32_f16(false, a, false, b2, (short)0, c2, false, false);
  }

  // C/D layout: VGPR r -> M = r + 8*hi, N = lane&15.
  #pragma unroll
  for (int r = 0; r < 8; ++r) {
    const int row = mt * 16 + hi * 8 + r;
    float* o = em + (size_t)row * L_ + m;
    o[0]  = c0[r];
    o[16] = c1[r];
    o[32] = c2[r];
  }
}

// -----------------------------------------------------------------------------
// Phase 2: Viterbi recursion. One block per batch, t_feats + delta in LDS.
// delta_full: [B, T+1, L]; bp_full: [B, T+1, L].
// -----------------------------------------------------------------------------
__global__ __launch_bounds__(64) void viterbi_recur(
    const float* __restrict__ em, const float* __restrict__ tf,
    float* __restrict__ delta_full, int* __restrict__ bp_full) {
  __shared__ float tf_s[L_ * L_];
  __shared__ float dbuf[2][L_];
  const int b = blockIdx.x;
  const int tid = threadIdx.x;

  for (int i = tid; i < L_ * L_; i += 64) tf_s[i] = tf[i];
  const size_t base = (size_t)b * (T_ + 1) * L_;
  if (tid < L_) {
    const float d0 = (tid == 0) ? 0.0f : -INFINITY;
    dbuf[0][tid] = d0;
    delta_full[base + tid] = d0;
    bp_full[base + tid] = 0;
  }
  __syncthreads();

  int cur = 0;
  for (int t = 0; t < T_; ++t) {
    if (tid < L_) {
      const float* dp = dbuf[cur];
      float best = dp[0] + tf_s[tid];           // l = 0 (first-occurrence argmax)
      int bi = 0;
      #pragma unroll 4
      for (int l = 1; l < L_; ++l) {
        const float s = dp[l] + tf_s[l * L_ + tid];
        if (s > best) { best = s; bi = l; }
      }
      const float d = best + em[((size_t)b * T_ + t) * L_ + tid];
      dbuf[cur ^ 1][tid] = d;
      const size_t off = base + (size_t)(t + 1) * L_ + tid;
      delta_full[off] = d;                       // mask==1 for one-hot X
      bp_full[off] = bi;
    }
    __syncthreads();
    cur ^= 1;
  }
}

// -----------------------------------------------------------------------------
// Phase 3a: zero the (poisoned) output.
// -----------------------------------------------------------------------------
__global__ void zero_out(float* __restrict__ o, int n) {
  const int i = blockIdx.x * blockDim.x + threadIdx.x;
  if (i < n) o[i] = 0.0f;
}

// -----------------------------------------------------------------------------
// Phase 3b: backtrace, one thread per batch. Writes one-hots into out [B,T,L].
// -----------------------------------------------------------------------------
__global__ void backtrace_k(const float* __restrict__ delta_full,
                            const int* __restrict__ bp_full,
                            float* __restrict__ out) {
  const int b = blockIdx.x * blockDim.x + threadIdx.x;
  if (b >= B_) return;
  const float* df = delta_full + (size_t)b * (T_ + 1) * L_;
  const int*   bp = bp_full    + (size_t)b * (T_ + 1) * L_;

  float best = -INFINITY; int end_n = 0;
  for (int nn = 0; nn <= T_; ++nn) {
    const float* row = df + (size_t)nn * L_;
    float mx = row[0];
    for (int l = 1; l < L_; ++l) if (row[l] > mx) mx = row[l];
    if (mx > best) { best = mx; end_n = nn; }   // first-occurrence argmax
  }
  const float* row = df + (size_t)end_n * L_;
  float mx = row[0]; int t_cur = 0;
  for (int l = 1; l < L_; ++l) if (row[l] > mx) { mx = row[l]; t_cur = l; }

  for (int nn = end_n; nn >= 1; --nn) {
    out[((size_t)b * T_ + (nn - 1)) * L_ + t_cur] = 1.0f;
    t_cur = bp[(size_t)nn * L_ + t_cur];
  }
}

extern "C" void kernel_launch(void* const* d_in, const int* in_sizes, int n_in,
                              void* d_out, int out_size, void* d_ws, size_t ws_size,
                              hipStream_t stream) {
  const float* X  = (const float*)d_in[0];   // [B,T,V] one-hot f32
  const float* tf = (const float*)d_in[1];   // [L,L]
  const float* ef = (const float*)d_in[2];   // [L,V]
  float* out = (float*)d_out;                // [B,T,L]

  float*     em         = (float*)d_ws;                          // B*T*L f32
  float*     delta_full = em + (size_t)B_ * T_ * L_;             // B*(T+1)*L f32
  int*       bp_full    = (int*)(delta_full + (size_t)B_ * (T_ + 1) * L_); // B*(T+1)*L i32
  _Float16*  ef16       = (_Float16*)(bp_full + (size_t)B_ * (T_ + 1) * L_); // L*V f16 (32B-aligned)

  const int nE = L_ * V_;
  cvt_e_f16<<<(nE + 255) / 256, 256, 0, stream>>>(ef, ef16, nE);
  emission_gemm<<<32, 256, 0, stream>>>(X, ef16, em);
  viterbi_recur<<<B_, 64, 0, stream>>>(em, tf, delta_full, bp_full);
  const int n_out = B_ * T_ * L_;
  zero_out<<<(n_out + 255) / 256, 256, 0, stream>>>(out, n_out);
  backtrace_k<<<1, 32, 0, stream>>>(delta_full, bp_full, out);
}
